// DeformableAttention1D_16673063043373
// MI455X (gfx1250) — compile-verified, matrix operand order checked by reading, not run on hardware
//
#include <hip/hip_runtime.h>
#include <hip/hip_bf16.h>

typedef __bf16 bf16_t;
typedef bf16_t v16bf  __attribute__((ext_vector_type(16)));
typedef bf16_t bf16x8 __attribute__((ext_vector_type(8)));
typedef float  v8f    __attribute__((ext_vector_type(8)));
typedef unsigned int u32x4 __attribute__((ext_vector_type(4)));
typedef int          i32x8 __attribute__((ext_vector_type(8)));
typedef int          i32x4 __attribute__((ext_vector_type(4)));

#define DM 512
#define NH 8
#define NP 4
#define HD 64

#if defined(__has_builtin)
#if __has_builtin(__builtin_amdgcn_tensor_load_to_lds) && \
    __has_builtin(__builtin_amdgcn_s_wait_tensorcnt)
#define USE_TDM 1
#endif
#endif

// ---------------- f32 -> bf16 conversion ----------------
__global__ void cvt_f32_to_bf16(const float* __restrict__ in,
                                bf16_t* __restrict__ out, int n) {
    int i = blockIdx.x * blockDim.x + threadIdx.x;
    int stride = gridDim.x * blockDim.x;
    for (; i < n; i += stride) out[i] = (bf16_t)in[i];
}

#ifdef USE_TDM
// Issue one TDM descriptor: 2D tile (rows x 32 bf16) from a (Mtot x Ktot)
// row-major bf16 tensor into LDS, padding +16B after each 64B row so the
// LDS image has an 80B (LDK=40) row stride. ISA 8.3/8.4 bit packing.
__device__ __forceinline__ void tdm_load_tile(const bf16_t* gptr,
                                              unsigned int lds_addr, int rows,
                                              int Ktot, int Mtot) {
    unsigned long long ga = (unsigned long long)(uintptr_t)gptr;
    u32x4 g0;
    g0.x = 1u;                                    // count=1, user descriptor
    g0.y = lds_addr;                              // [63:32] lds_addr (bytes)
    g0.z = (unsigned int)(ga & 0xFFFFFFFFu);      // [95:64] global_addr lo
    g0.w = (unsigned int)((ga >> 32) & 0x01FFFFFFu) | (2u << 30);  // addr hi + type=2
    i32x8 g1;
    // data_size=1(2B), pad_enable, pad_interval=3(16 dw), pad_amount=3(4 dw)
    g1[0] = (int)((1u << 16) | (1u << 20) | (3u << 22) | (3u << 25));
    g1[1] = (int)(((unsigned)Ktot & 0xFFFFu) << 16);                 // tensor_dim0 lo
    g1[2] = (int)((((unsigned)Ktot >> 16) & 0xFFFFu) |
                  (((unsigned)Mtot & 0xFFFFu) << 16));               // dim0 hi | dim1 lo
    g1[3] = (int)((((unsigned)Mtot >> 16) & 0xFFFFu) | (32u << 16)); // dim1 hi | tile_dim0=32
    g1[4] = (int)((unsigned)rows & 0xFFFFu);                         // tile_dim1
    g1[5] = Ktot;                                                    // tensor_dim0_stride lo
    g1[6] = 0;
    g1[7] = 0;
    i32x4 z4 = {0, 0, 0, 0};
#if __clang_major__ >= 23
    i32x8 z8 = {0, 0, 0, 0, 0, 0, 0, 0};
    __builtin_amdgcn_tensor_load_to_lds(g0, g1, z4, z4, z8, 0);
#else
    __builtin_amdgcn_tensor_load_to_lds(g0, g1, z4, z4, 0);
#endif
}
#endif  // USE_TDM

// ---------------- WMMA GEMM: C = A(MxK) * W(NxK)^T + bias ----------------
union FragU { v16bf v; bf16x8 h[2]; };

// ISA 7.12.2: 16-bit A 16x32 fragment. lane&15 = row, lane>>4 selects K-half.
__device__ __forceinline__ v16bf load_frag_row(const bf16_t* __restrict__ row,
                                               int khalf) {
    FragU u;
    u.h[0] = *(const bf16x8*)(row + khalf * 8);
    u.h[1] = *(const bf16x8*)(row + 16 + khalf * 8);
    return u.v;
}

// Block = WMW*WNW wave32s; each wave owns a 64x32 output tile (4x2 WMMA frags).
// KD compile-time; BK = 32 per step; LDS tiles double-buffered for the TDM path.
template <int BM, int BN, int WMW, int WNW, int KD>
__global__ __launch_bounds__(WMW * WNW * 32) void wmma_gemm_bf16(
    const bf16_t* __restrict__ A, const bf16_t* __restrict__ W,
    const float* __restrict__ bias, float* __restrict__ C, int M, int N) {
    constexpr int NT  = WMW * WNW * 32;
    constexpr int LDK = 40;  // 80B row stride: 16B aligned, staggers banks
    __shared__ __align__(16) bf16_t As[2][BM][LDK];
    __shared__ __align__(16) bf16_t Bs[2][BN][LDK];

    const int tid    = threadIdx.x;
    const int lane   = tid & 31;
    const int wid    = tid >> 5;
    const int wave_m = wid % WMW;
    const int wave_n = wid / WMW;
    const int bm0    = blockIdx.y * BM;
    const int bn0    = blockIdx.x * BN;
    const int khalf  = lane >> 4;
    const int lrow   = lane & 15;

    v8f acc[4][2] = {};

    auto compute = [&](int buf) {
        v16bf bfr[2];
#pragma unroll
        for (int j = 0; j < 2; ++j)
            bfr[j] = load_frag_row(&Bs[buf][wave_n * 32 + j * 16 + lrow][0], khalf);
#pragma unroll
        for (int i = 0; i < 4; ++i) {
            v16bf afr = load_frag_row(&As[buf][wave_m * 64 + i * 16 + lrow][0], khalf);
#pragma unroll
            for (int j = 0; j < 2; ++j)
                acc[i][j] = __builtin_amdgcn_wmma_f32_16x16x32_bf16(
                    false, afr, false, bfr[j], (short)0, acc[i][j], false, false);
        }
    };

#ifdef USE_TDM
    // ---- Tensor Data Mover staging, double buffered ----
    if (wid == 0) {
        tdm_load_tile(A + (size_t)bm0 * KD,
                      (unsigned int)(uintptr_t)&As[0][0][0], BM, KD, M);
        tdm_load_tile(W + (size_t)bn0 * KD,
                      (unsigned int)(uintptr_t)&Bs[0][0][0], BN, KD, N);
    }
#pragma unroll 4
    for (int kk = 0; kk < KD; kk += 32) {
        const int cur = (kk >> 5) & 1;
        if (wid == 0) __builtin_amdgcn_s_wait_tensorcnt(0);  // tile `cur` landed
        __syncthreads();
        if (wid == 0 && kk + 32 < KD) {  // prefetch next tile into other buffer
            tdm_load_tile(A + (size_t)bm0 * KD + (kk + 32),
                          (unsigned int)(uintptr_t)&As[cur ^ 1][0][0], BM, KD, M);
            tdm_load_tile(W + (size_t)bn0 * KD + (kk + 32),
                          (unsigned int)(uintptr_t)&Bs[cur ^ 1][0][0], BN, KD, N);
        }
        compute(cur);
        __syncthreads();  // everyone done reading `cur` before it is re-filled
    }
#else
    // ---- fallback: register-staged b128 copies (fully unrolled, KD constexpr) ----
    for (int kk = 0; kk < KD; kk += 32) {
        __syncthreads();
        constexpr int AV = BM * 4;
#pragma unroll
        for (int it = 0; it < (AV + NT - 1) / NT; ++it) {
            int v = tid + it * NT;
            if ((AV % NT) && v >= AV) break;
            int r = v >> 2, kc = (v & 3) * 8;
            *(bf16x8*)&As[0][r][kc] =
                *(const bf16x8*)&A[(size_t)(bm0 + r) * KD + kk + kc];
        }
        constexpr int BV = BN * 4;
#pragma unroll
        for (int it = 0; it < (BV + NT - 1) / NT; ++it) {
            int v = tid + it * NT;
            if ((BV % NT) && v >= BV) break;
            int r = v >> 2, kc = (v & 3) * 8;
            *(bf16x8*)&Bs[0][r][kc] =
                *(const bf16x8*)&W[(size_t)(bn0 + r) * KD + kk + kc];
        }
        if (kk + 32 < KD) {  // gfx1250 global_prefetch_b8 hint
            int pr = tid >> 1;
            if (pr < BM) __builtin_prefetch(&A[(size_t)(bm0 + pr) * KD + kk + 32], 0, 1);
            if (pr < BN) __builtin_prefetch(&W[(size_t)(bn0 + pr) * KD + kk + 32], 0, 1);
        }
        __syncthreads();
        compute(0);
    }
#endif

    // C/D layout (ISA 7.12.2): VGPR r -> M = r + 8*(lane>>4), N = lane&15.
    const int nloc  = lrow;
    const int mbase = khalf * 8;
#pragma unroll
    for (int i = 0; i < 4; ++i) {
#pragma unroll
        for (int j = 0; j < 2; ++j) {
            int gn  = bn0 + wave_n * 32 + j * 16 + nloc;
            float b = bias[gn];
#pragma unroll
            for (int r = 0; r < 8; ++r) {
                int gm = bm0 + wave_m * 64 + i * 16 + mbase + r;
                C[(size_t)gm * N + gn] = acc[i][j][r] + b;
            }
        }
    }
}

// ---------------- deformable gather + softmax + weighted sum ----------------
// One block per (b,l); one wave per head h. Lane covers d and d+32.
__global__ __launch_bounds__(256) void deform_attn_kernel(
    const float* __restrict__ q, const float* __restrict__ kmat,
    const float* __restrict__ vmat, const float* __restrict__ offs,
    const float* __restrict__ lgts, bf16_t* __restrict__ attn_bf,
    int Bn, int L) {
    const int bl   = blockIdx.x;  // b*L + l
    const int b    = bl / L;
    const int l    = bl - b * L;
    const int tid  = threadIdx.x;
    const int h    = tid >> 5;
    const int lane = tid & 31;

    const size_t rowq = (size_t)bl * DM + h * HD;
    const float q0 = q[rowq + lane];
    const float q1 = q[rowq + lane + 32];
    const size_t rowp = (size_t)bl * (NH * NP) + h * NP;

    float frac[NP];
    int   idx0[NP], idx1[NP];
    float score[NP];

#pragma unroll
    for (int p = 0; p < NP; ++p) {
        float pos = (float)l + offs[rowp + p];
        pos = fminf(fmaxf(pos, 0.0f), (float)(L - 1));
        int i0 = (int)floorf(pos);
        if (i0 > L - 1) i0 = L - 1;
        int i1 = i0 + 1;
        if (i1 > L - 1) i1 = L - 1;
        float w = pos - (float)i0;
        idx0[p] = i0; idx1[p] = i1; frac[p] = w;

        const size_t r0 = ((size_t)(b * L + i0)) * DM + h * HD;
        const size_t r1 = ((size_t)(b * L + i1)) * DM + h * HD;
        float k00 = kmat[r0 + lane], k01 = kmat[r0 + lane + 32];
        float k10 = kmat[r1 + lane], k11 = kmat[r1 + lane + 32];
        float ks0 = k00 + w * (k10 - k00);
        float ks1 = k01 + w * (k11 - k01);
        float d = q0 * ks0 + q1 * ks1;
#pragma unroll
        for (int s = 16; s > 0; s >>= 1) d += __shfl_xor(d, s, 32);
        score[p] = d * 0.125f + lgts[rowp + p];  // 1/sqrt(64) = 0.125
    }

    float m = score[0];
#pragma unroll
    for (int p = 1; p < NP; ++p) m = fmaxf(m, score[p]);
    float e[NP], sum = 0.0f;
#pragma unroll
    for (int p = 0; p < NP; ++p) { e[p] = __expf(score[p] - m); sum += e[p]; }
    const float inv = 1.0f / sum;

    float o0 = 0.0f, o1 = 0.0f;
#pragma unroll
    for (int p = 0; p < NP; ++p) {
        const float wp = e[p] * inv;
        const float w  = frac[p];
        const size_t r0 = ((size_t)(b * L + idx0[p])) * DM + h * HD;
        const size_t r1 = ((size_t)(b * L + idx1[p])) * DM + h * HD;
        float v00 = vmat[r0 + lane], v01 = vmat[r0 + lane + 32];
        float v10 = vmat[r1 + lane], v11 = vmat[r1 + lane + 32];
        o0 += wp * (v00 + w * (v10 - v00));
        o1 += wp * (v01 + w * (v11 - v01));
    }
    attn_bf[rowq + lane]      = (bf16_t)o0;
    attn_bf[rowq + lane + 32] = (bf16_t)o1;
}

// ---------------- host-side orchestration ----------------
extern "C" void kernel_launch(void* const* d_in, const int* in_sizes, int n_in,
                              void* d_out, int out_size, void* d_ws,
                              size_t ws_size, hipStream_t stream) {
    const float* q_in  = (const float*)d_in[0];
    const float* kv_in = (const float*)d_in[1];
    const float* Wq = (const float*)d_in[2];  const float* bq = (const float*)d_in[3];
    const float* Wk = (const float*)d_in[4];  const float* bk = (const float*)d_in[5];
    const float* Wv = (const float*)d_in[6];  const float* bv = (const float*)d_in[7];
    const float* Woff = (const float*)d_in[8];  const float* boff = (const float*)d_in[9];
    const float* Wa = (const float*)d_in[10]; const float* ba = (const float*)d_in[11];
    const float* Wo = (const float*)d_in[12]; const float* bo = (const float*)d_in[13];
    float* out = (float*)d_out;

    const int Bn = 4, L = 4096;
    const int M  = Bn * L;     // 16384 rows
    const int K  = DM;         // 512
    const int Ns = NH * NP;    // 32

    char* ws = (char*)d_ws;
    size_t off = 0;
    auto alloc = [&](size_t bytes) -> void* {
        void* p = ws + off;
        off = (off + bytes + 255) & ~(size_t)255;
        return p;
    };

    bf16_t* qbf    = (bf16_t*)alloc((size_t)M * K * 2);
    bf16_t* kvbf   = (bf16_t*)alloc((size_t)M * K * 2);
    bf16_t* Wqbf   = (bf16_t*)alloc((size_t)K * K * 2);
    bf16_t* Wkbf   = (bf16_t*)alloc((size_t)K * K * 2);
    bf16_t* Wvbf   = (bf16_t*)alloc((size_t)K * K * 2);
    bf16_t* Wobf   = (bf16_t*)alloc((size_t)K * K * 2);
    bf16_t* Woffbf = (bf16_t*)alloc((size_t)Ns * K * 2);
    bf16_t* Wabf   = (bf16_t*)alloc((size_t)Ns * K * 2);
    float*  qf     = (float*)alloc((size_t)M * K * 4);
    float*  kf     = (float*)alloc((size_t)M * K * 4);
    float*  vf     = (float*)alloc((size_t)M * K * 4);
    float*  offsf  = (float*)alloc((size_t)M * Ns * 4);
    float*  lgtsf  = (float*)alloc((size_t)M * Ns * 4);
    bf16_t* attnbf = (bf16_t*)alloc((size_t)M * K * 2);

    auto cvt = [&](const float* src, bf16_t* dst, int n) {
        int grid = (n + 255) / 256;
        if (grid > 4096) grid = 4096;
        cvt_f32_to_bf16<<<grid, 256, 0, stream>>>(src, dst, n);
    };
    cvt(q_in,  qbf,  M * K);
    cvt(kv_in, kvbf, M * K);
    cvt(Wq, Wqbf, K * K);
    cvt(Wk, Wkbf, K * K);
    cvt(Wv, Wvbf, K * K);
    cvt(Wo, Wobf, K * K);
    cvt(Woff, Woffbf, Ns * K);
    cvt(Wa,   Wabf,   Ns * K);

    // Big projections: 128x128 block tile, 2x4 waves (256 threads).
    dim3 gridBig(K / 128, M / 128);
    wmma_gemm_bf16<128, 128, 2, 4, DM><<<gridBig, 256, 0, stream>>>(qbf,  Wqbf, bq, qf, M, K);
    wmma_gemm_bf16<128, 128, 2, 4, DM><<<gridBig, 256, 0, stream>>>(kvbf, Wkbf, bk, kf, M, K);
    wmma_gemm_bf16<128, 128, 2, 4, DM><<<gridBig, 256, 0, stream>>>(kvbf, Wvbf, bv, vf, M, K);

    // Narrow projections (N=32): 256x32 block tile, 4x1 waves (128 threads).
    dim3 gridSm(1, M / 256);
    wmma_gemm_bf16<256, 32, 4, 1, DM><<<gridSm, 128, 0, stream>>>(qbf, Woffbf, boff, offsf, M, Ns);
    wmma_gemm_bf16<256, 32, 4, 1, DM><<<gridSm, 128, 0, stream>>>(qbf, Wabf,   ba,   lgtsf, M, Ns);

    // Deformable gather / softmax / weighted sum -> bf16 rows (head-major).
    deform_attn_kernel<<<M, 256, 0, stream>>>(qf, kf, vf, offsf, lgtsf, attnbf, Bn, L);

    // Output projection straight into d_out (f32).
    wmma_gemm_bf16<128, 128, 2, 4, DM><<<gridBig, 256, 0, stream>>>(attnbf, Wobf, bo, out, M, K);
}
